// ConvGraphLayer_32341103738940
// MI455X (gfx1250) — matrix operand order repaired and checked by reading, not compiled, not run on hardware
//
#include <hip/hip_runtime.h>

// ConvGraphLayer for MI455X (gfx1250, wave32, WMMA bf16)
//
// out = relu([x | (adj@x)/(nn+eps)] @ fc_w^T + fc_b)
// Refactored:  Z = x@W2^T (bf16, transposed, K-padded, L2-resident)
//              out = relu( x@W1^T + D*(adj@Z) + b ),  D = diag(1/(nn+eps))
// adj (400 MB fp32) is streamed exactly once with nontemporal loads ->
// HBM-bound at ~17 us; bf16 WMMA supplies the required ~0.75 PFLOP/s.

#define NNODES 10000
#define FDIM   64
#define HDIM   64
#define KC     128                 // K-chunk staged per iteration
#define NPAD   10112               // 79 * 128 (K padded, zero-filled tail)
#define ITERS  (NPAD / KC)         // 79
#define EPSV   1e-7f

typedef __attribute__((ext_vector_type(16))) __bf16 v16bf;
typedef __attribute__((ext_vector_type(8)))  __bf16 v8bf;
typedef __attribute__((ext_vector_type(8)))  float  v8f;
typedef __attribute__((ext_vector_type(4)))  float  v4f;

__device__ __forceinline__ __bf16 bf(float f) { return (__bf16)f; }

// ---------------------------------------------------------------------------
// Kernel 1: Zt[h][i] = sum_f x[i,f] * fc_w[h, 64+f]  (bf16, transposed),
// plus zero-fill of the K-pad region [10000, NPAD).
// One wave per 16-row tile of x; 4 n-tiles (h) x 2 k-chunks -> 8 WMMAs.
// ---------------------------------------------------------------------------
__global__ __launch_bounds__(32) void convgraph_zt(
    const float* __restrict__ x, const float* __restrict__ fc_w,
    __bf16* __restrict__ zt) {
  const int row0 = blockIdx.x * 16;
  const int lane = threadIdx.x & 31;
  const int lh   = lane & 15;
  const int hi   = lane >> 4;

  // A fragment: 16x64 x-tile (M=lh). ISA layout: lanes<16 hold K {0-7,16-23},
  // lanes>=16 hold K {8-15,24-31} per 32-wide chunk.
  const float* xr = x + (size_t)(row0 + lh) * FDIM;
  v16bf a[2];
#pragma unroll
  for (int c = 0; c < 2; ++c)
#pragma unroll
    for (int e = 0; e < 16; ++e)
      a[c][e] = bf(xr[c * 32 + 8 * hi + (e < 8 ? e : e + 8)]);

#pragma unroll
  for (int nt = 0; nt < 4; ++nt) {
    // B[k][n] = W2[16*nt+n, k] = fc_w[(16*nt+n)*128 + 64 + k]
    const float* wr = fc_w + (size_t)(16 * nt + lh) * (2 * FDIM) + FDIM;
    v8f acc = {};
#pragma unroll
    for (int c = 0; c < 2; ++c) {
      v16bf b;
#pragma unroll
      for (int e = 0; e < 16; ++e) b[e] = bf(wr[c * 32 + 8 * hi + (e < 8 ? e : e + 8)]);
      acc = __builtin_amdgcn_wmma_f32_16x16x32_bf16(false, a[c], false, b,
                                                    (short)0, acc, false, false);
    }
    // D element (m,n): m = r + 8*hi, n = lh  ->  Zt[16*nt+n][row0+m]
    __bf16* zc = zt + (size_t)(16 * nt + lh) * NPAD + row0 + 8 * hi;
#pragma unroll
    for (int r = 0; r < 8; ++r) zc[r] = bf(acc[r]);
  }

  // zero-fill K pad so the main GEMM tail adds zeros
  if (blockIdx.x < HDIM) {
    __bf16* zp = zt + (size_t)blockIdx.x * NPAD + NNODES;
    for (int e = lane; e < NPAD - NNODES; e += 32) zp[e] = bf(0.0f);
  }
}

// ---------------------------------------------------------------------------
// Main kernel: one block (4 waves) per 16-row tile of adj.
//   acc = adj_tile(16 x K) @ Zt(K x 64)   [wave w owns output cols 16w..16w+15]
//   acc = acc / (nn+eps)  +  x_tile @ W1^T  + bias, relu, store fp32.
// adj is staged fp32->bf16 into double-buffered LDS (one convert per element,
// shared by all 4 waves); adj loads are nontemporal to protect L2 (Zt lives
// there and is reused by all 625 blocks).
// ---------------------------------------------------------------------------
__global__ __launch_bounds__(128) void convgraph_main(
    const float* __restrict__ adj, const float* __restrict__ x,
    const float* __restrict__ nn, const float* __restrict__ fc_w,
    const float* __restrict__ fc_b, const __bf16* __restrict__ zt,
    float* __restrict__ out) {
  // +8 bf16 row pad: per-lane b128 fragment reads hit disjoint bank groups
  __shared__ alignas(32) __bf16 As[2][16][KC + 8];

  const int row0 = blockIdx.x * 16;
  const int t    = threadIdx.x;
  const int wave = t >> 5;
  const int lane = t & 31;
  const int lh   = lane & 15;
  const int hi   = lane >> 4;

  // staging map: thread -> (row ms, 16-float column block ks)
  const int ms = t >> 3;
  const int ks = (t & 7) * 16;
  const float* adj_row = adj + (size_t)(row0 + ms) * NNODES;

  v4f st[4];
  auto stage_load = [&](int it) {
    const int k0 = it * KC + ks;
    if (k0 + 15 < NNODES) {
#pragma unroll
      for (int q = 0; q < 4; ++q)
        st[q] = __builtin_nontemporal_load((const v4f*)(adj_row + k0 + 4 * q));
    } else {
#pragma unroll
      for (int q = 0; q < 4; ++q)
#pragma unroll
        for (int e = 0; e < 4; ++e) {
          const int k = k0 + 4 * q + e;
          st[q][e] = (k < NNODES) ? adj_row[k] : 0.0f;
        }
    }
  };
  auto stage_store = [&](int buf) {
#pragma unroll
    for (int q = 0; q < 2; ++q) {
      v8bf v;
#pragma unroll
      for (int e = 0; e < 4; ++e) v[e]     = bf(st[2 * q][e]);
#pragma unroll
      for (int e = 0; e < 4; ++e) v[4 + e] = bf(st[2 * q + 1][e]);
      *(v8bf*)&As[buf][ms][ks + 8 * q] = v;
    }
  };

  // B source: Zt rows are contiguous in K -> 32B contiguous per lane
  const __bf16* ztw = zt + (size_t)(16 * wave + lh) * NPAD + 16 * hi;

  stage_load(0);
  stage_store(0);
  __syncthreads();

  v8f acc = {};
  for (int it = 0; it < ITERS; ++it) {
    const int  cur  = it & 1;
    const bool more = (it + 1 < ITERS);
    if (more) stage_load(it + 1);           // prefetch next chunk into regs
#pragma unroll
    for (int c = 0; c < KC / 32; ++c) {
      union { v16bf v; v8bf h[2]; } ua;
      const __bf16* p = &As[cur][lh][c * 32 + hi * 16];
      ua.h[0] = *(const v8bf*)p;            // K {0-7}  (or {16-23} for hi lanes)
      ua.h[1] = *(const v8bf*)(p + 8);      // K {8-15} (or {24-31})
      const v16bf b = *(const v16bf*)(ztw + it * KC + c * 32);
      acc = __builtin_amdgcn_wmma_f32_16x16x32_bf16(false, ua.v, false, b,
                                                    (short)0, acc, false, false);
    }
    if (more) stage_store(1 - cur);         // fill the other buffer
    __syncthreads();                        // publish stores / retire reads
  }

  // per-row neighbor-count normalization (row m = r + 8*hi of this tile)
#pragma unroll
  for (int r = 0; r < 8; ++r)
    acc[r] *= 1.0f / (nn[row0 + r + 8 * hi] + EPSV);

  // + x_tile @ W1^T for this wave's 16 output columns
  const float* xr = x + (size_t)(row0 + lh) * FDIM;
  const float* wr = fc_w + (size_t)(16 * wave + lh) * (2 * FDIM);
#pragma unroll
  for (int c = 0; c < 2; ++c) {
    v16bf a, b;
#pragma unroll
    for (int e = 0; e < 16; ++e)
      a[e] = bf(xr[c * 32 + 8 * hi + (e < 8 ? e : e + 8)]);
#pragma unroll
    for (int e = 0; e < 16; ++e) b[e] = bf(wr[c * 32 + 8 * hi + (e < 8 ? e : e + 8)]);
    acc = __builtin_amdgcn_wmma_f32_16x16x32_bf16(false, a, false, b,
                                                  (short)0, acc, false, false);
  }

  const float bias = fc_b[16 * wave + lh];
#pragma unroll
  for (int r = 0; r < 8; ++r) {
    const float v = acc[r] + bias;
    out[(size_t)(row0 + r + 8 * hi) * HDIM + 16 * wave + lh] =
        v > 0.0f ? v : 0.0f;
  }
}

// ---------------------------------------------------------------------------
extern "C" void kernel_launch(void* const* d_in, const int* in_sizes, int n_in,
                              void* d_out, int out_size, void* d_ws, size_t ws_size,
                              hipStream_t stream) {
  (void)in_sizes; (void)n_in; (void)out_size; (void)ws_size;
  const float* x   = (const float*)d_in[0];
  const float* adj = (const float*)d_in[1];
  const float* nn  = (const float*)d_in[2];
  const float* fcw = (const float*)d_in[3];
  const float* fcb = (const float*)d_in[4];
  float*  out = (float*)d_out;
  __bf16* zt  = (__bf16*)d_ws;              // 64 * 10112 * 2 B = 1.24 MiB

  convgraph_zt  <<<NNODES / 16,  32, 0, stream>>>(x, fcw, zt);
  convgraph_main<<<NNODES / 16, 128, 0, stream>>>(adj, x, nn, fcw, fcb, zt, out);
}
